// CrossModalGenerator_37022618091615
// MI455X (gfx1250) — matrix operand order repaired and checked by reading, not compiled, not run on hardware
//
#include <hip/hip_runtime.h>
#include <cmath>

typedef __attribute__((ext_vector_type(16))) _Float16 v16h;
typedef __attribute__((ext_vector_type(8)))  _Float16 v8h;
typedef __attribute__((ext_vector_type(8)))  float    v8f;

#define BATCH 32768
#define D_IMG 768
#define D_TXT 512
#define FH    256

__device__ __forceinline__ float gelu_exact(float x) {
    return 0.5f * x * (1.0f + erff(x * 0.70710678118654752f));
}

// ---------------------------------------------------------------------------
// Weight convert: W [K,N] f32 (row-major) -> WT [N,K] f16 (row-major)
// ---------------------------------------------------------------------------
__global__ void wt_convert_kernel(const float* __restrict__ W,
                                  _Float16* __restrict__ WT, int K, int N) {
    int idx = blockIdx.x * 256 + threadIdx.x;
    if (idx < K * N) {
        int k = idx / N, n = idx % N;
        WT[(size_t)n * K + k] = (_Float16)W[idx];
    }
}

// ---------------------------------------------------------------------------
// Prior hidden: h = gelu(LN(noise[b]*w1 + b1)); one row (256 cols) per block
// ---------------------------------------------------------------------------
__global__ __launch_bounds__(256)
void prior_kernel(const float* __restrict__ noise, const float* __restrict__ w1,
                  const float* __restrict__ b1, const float* __restrict__ g,
                  const float* __restrict__ be, _Float16* __restrict__ H) {
    __shared__ float2 red[256];
    const int b = blockIdx.x, j = threadIdx.x;
    float x = noise[b] * w1[j] + b1[j];
    red[j] = make_float2(x, x * x);
    __syncthreads();
    for (int s = 128; s > 0; s >>= 1) {
        if (j < s) { red[j].x += red[j + s].x; red[j].y += red[j + s].y; }
        __syncthreads();
    }
    float mu  = red[0].x * (1.0f / 256.0f);
    float var = red[0].y * (1.0f / 256.0f) - mu * mu;
    float y = g[j] * (x - mu) * rsqrtf(var + 1e-5f) + be[j];
    H[(size_t)b * 256 + j] = (_Float16)gelu_exact(y);
}

// ---------------------------------------------------------------------------
// Fused GEMM + bias + LayerNorm + GELU.
// A: [B,K] (f32 or f16), WT: [256,K] f16 (transposed weights), H out: [B,256] f16
// Block: 256 threads (8 waves). WG tile M=64 x N=256. Wave tile M32 x N64.
// ---------------------------------------------------------------------------
template <bool AF32>
__global__ __launch_bounds__(256)
void gemm_ln_gelu_kernel(const void* __restrict__ Aptr, int K,
                         const _Float16* __restrict__ WT,
                         const float* __restrict__ bias,
                         const float* __restrict__ gam,
                         const float* __restrict__ bet,
                         _Float16* __restrict__ H) {
    constexpr int KSTEP = 64, LDA = KSTEP + 8;
    __shared__ _Float16 ldsA[64 * LDA];
    __shared__ float2   part[64][4];
    __shared__ float2   musig[64];

    const int tid  = threadIdx.x;
    const int lane = tid & 31, wave = tid >> 5;
    const int Mt = wave >> 2, Nt = wave & 3;
    const int mlo = lane & 15, kh = lane >> 4;
    const int rowbase = blockIdx.x * 64;
    const int srow = tid >> 2, sq = tid & 3;

    v8f acc[2][4] = {};

    const int ksteps = K / KSTEP;
    for (int step = 0; step < ksteps; ++step) {
        __syncthreads();
        {   // stage A block 64 x 64 into LDS as f16
            const int kofs = step * KSTEP + sq * 16;
            _Float16* dst = &ldsA[srow * LDA + sq * 16];
            if (AF32) {
                const float* src = (const float*)Aptr + (size_t)(rowbase + srow) * K + kofs;
#pragma unroll
                for (int i = 0; i < 16; ++i) dst[i] = (_Float16)src[i];
            } else {
                const _Float16* src = (const _Float16*)Aptr + (size_t)(rowbase + srow) * K + kofs;
#pragma unroll
                for (int i = 0; i < 16; ++i) dst[i] = src[i];
            }
        }
        __syncthreads();
#pragma unroll
        for (int kk = 0; kk < KSTEP; kk += 32) {
            v16h af[2];
#pragma unroll
            for (int mi = 0; mi < 2; ++mi) {
                const int rowL = Mt * 32 + mi * 16 + mlo;
                const _Float16* p = &ldsA[rowL * LDA + kk + kh * 16];
                v8h lo = *(const v8h*)p;
                v8h hi = *(const v8h*)(p + 8);
                af[mi] = __builtin_shufflevector(lo, hi, 0,1,2,3,4,5,6,7,8,9,10,11,12,13,14,15);
            }
#pragma unroll
            for (int ni = 0; ni < 4; ++ni) {
                const int col = Nt * 64 + ni * 16 + mlo;
                const _Float16* bp = WT + (size_t)col * K + step * KSTEP + kk + kh * 16;
                if (kk == 0) __builtin_prefetch(bp + KSTEP, 0, 0);
                v16h bf = *(const v16h*)bp;
#pragma unroll
                for (int mi = 0; mi < 2; ++mi)
                    acc[mi][ni] = __builtin_amdgcn_wmma_f32_16x16x32_f16(
                        false, af[mi], false, bf, (short)0, acc[mi][ni], false, false);
            }
        }
    }

    // ---- epilogue: bias, LN (row-wise over 256 cols), gelu, store f16 ----
    float bch[4], gch[4], bech[4];
#pragma unroll
    for (int ni = 0; ni < 4; ++ni) {
        const int col = Nt * 64 + ni * 16 + mlo;
        bch[ni] = bias[col]; gch[ni] = gam[col]; bech[ni] = bet[col];
    }
#pragma unroll
    for (int mi = 0; mi < 2; ++mi)
#pragma unroll
        for (int ni = 0; ni < 4; ++ni)
#pragma unroll
            for (int v = 0; v < 8; ++v) acc[mi][ni][v] += bch[ni];

#pragma unroll
    for (int mi = 0; mi < 2; ++mi) {
#pragma unroll
        for (int v = 0; v < 8; ++v) {
            float s = 0.f, q = 0.f;
#pragma unroll
            for (int ni = 0; ni < 4; ++ni) { float x = acc[mi][ni][v]; s += x; q += x * x; }
#pragma unroll
            for (int m = 8; m >= 1; m >>= 1) {
                s += __shfl_xor(s, m, 32);
                q += __shfl_xor(q, m, 32);
            }
            if (mlo == 0) {
                int rowL = Mt * 32 + mi * 16 + kh * 8 + v;
                part[rowL][Nt] = make_float2(s, q);
            }
        }
    }
    __syncthreads();
    if (tid < 64) {
        float s = 0.f, q = 0.f;
        for (int w = 0; w < 4; ++w) { s += part[tid][w].x; q += part[tid][w].y; }
        float mu  = s * (1.0f / 256.0f);
        float var = q * (1.0f / 256.0f) - mu * mu;
        musig[tid] = make_float2(mu, rsqrtf(var + 1e-5f));
    }
    __syncthreads();
#pragma unroll
    for (int mi = 0; mi < 2; ++mi) {
#pragma unroll
        for (int v = 0; v < 8; ++v) {
            int rowL = Mt * 32 + mi * 16 + kh * 8 + v;
            float2 ms = musig[rowL];
            size_t rg = (size_t)(rowbase + rowL) * 256;
#pragma unroll
            for (int ni = 0; ni < 4; ++ni) {
                const int col = Nt * 64 + ni * 16 + mlo;
                float y = gch[ni] * (acc[mi][ni][v] - ms.x) * ms.y + bech[ni];
                H[rg + col] = (_Float16)gelu_exact(y);
            }
        }
    }
}

// ---------------------------------------------------------------------------
// Final stage: gen = tanh(A1@W1^T + b1), pri = tanh(A2@W2^T + b2),
// out = (mt==selGen)?gen : (mt==3)?pri : passthru.  K=256 fixed.
// Grid: (B/64, NOUT/256). Block 256 threads.
// ---------------------------------------------------------------------------
__global__ __launch_bounds__(256)
void gemm2_tanh_sel_kernel(const _Float16* __restrict__ A1, const _Float16* __restrict__ WT1,
                           const float* __restrict__ B1,
                           const _Float16* __restrict__ A2, const _Float16* __restrict__ WT2,
                           const float* __restrict__ B2,
                           const float* __restrict__ pass, const int* __restrict__ mt,
                           float* __restrict__ out, int NOUT, int selGen) {
    constexpr int K = 256, KSTEP = 64, LDA = KSTEP + 8;
    __shared__ _Float16 ldsA1[64 * LDA];
    __shared__ _Float16 ldsA2[64 * LDA];

    const int tid  = threadIdx.x;
    const int lane = tid & 31, wave = tid >> 5;
    const int Mt = wave >> 2, Nt = wave & 3;
    const int mlo = lane & 15, kh = lane >> 4;
    const int rowbase = blockIdx.x * 64;
    const int nbase   = blockIdx.y * 256;
    const int srow = tid >> 2, sq = tid & 3;

    v8f acc1[2][4] = {};
    v8f acc2[2][4] = {};

    for (int step = 0; step < K / KSTEP; ++step) {
        __syncthreads();
        {
            const int kofs = step * KSTEP + sq * 16;
            const _Float16* s1 = A1 + (size_t)(rowbase + srow) * K + kofs;
            const _Float16* s2 = A2 + (size_t)(rowbase + srow) * K + kofs;
            _Float16* d1 = &ldsA1[srow * LDA + sq * 16];
            _Float16* d2 = &ldsA2[srow * LDA + sq * 16];
#pragma unroll
            for (int i = 0; i < 16; ++i) { d1[i] = s1[i]; d2[i] = s2[i]; }
        }
        __syncthreads();
#pragma unroll
        for (int kk = 0; kk < KSTEP; kk += 32) {
            v16h af1[2], af2[2];
#pragma unroll
            for (int mi = 0; mi < 2; ++mi) {
                const int rowL = Mt * 32 + mi * 16 + mlo;
                const _Float16* p1 = &ldsA1[rowL * LDA + kk + kh * 16];
                const _Float16* p2 = &ldsA2[rowL * LDA + kk + kh * 16];
                v8h lo1 = *(const v8h*)p1, hi1 = *(const v8h*)(p1 + 8);
                v8h lo2 = *(const v8h*)p2, hi2 = *(const v8h*)(p2 + 8);
                af1[mi] = __builtin_shufflevector(lo1, hi1, 0,1,2,3,4,5,6,7,8,9,10,11,12,13,14,15);
                af2[mi] = __builtin_shufflevector(lo2, hi2, 0,1,2,3,4,5,6,7,8,9,10,11,12,13,14,15);
            }
#pragma unroll
            for (int ni = 0; ni < 4; ++ni) {
                const int col = nbase + Nt * 64 + ni * 16 + mlo;
                const _Float16* bp1 = WT1 + (size_t)col * K + step * KSTEP + kk + kh * 16;
                const _Float16* bp2 = WT2 + (size_t)col * K + step * KSTEP + kk + kh * 16;
                if (kk == 0) { __builtin_prefetch(bp1 + KSTEP, 0, 0); __builtin_prefetch(bp2 + KSTEP, 0, 0); }
                v16h bf1 = *(const v16h*)bp1;
                v16h bf2 = *(const v16h*)bp2;
#pragma unroll
                for (int mi = 0; mi < 2; ++mi) {
                    acc1[mi][ni] = __builtin_amdgcn_wmma_f32_16x16x32_f16(
                        false, af1[mi], false, bf1, (short)0, acc1[mi][ni], false, false);
                    acc2[mi][ni] = __builtin_amdgcn_wmma_f32_16x16x32_f16(
                        false, af2[mi], false, bf2, (short)0, acc2[mi][ni], false, false);
                }
            }
        }
    }

    // ---- epilogue: tanh + row select, write once to global ----
    float b1c[4], b2c[4];
#pragma unroll
    for (int ni = 0; ni < 4; ++ni) {
        const int col = nbase + Nt * 64 + ni * 16 + mlo;
        b1c[ni] = B1[col]; b2c[ni] = B2[col];
    }
#pragma unroll
    for (int mi = 0; mi < 2; ++mi) {
#pragma unroll
        for (int v = 0; v < 8; ++v) {
            const int row = rowbase + Mt * 32 + mi * 16 + kh * 8 + v;
            const int m = mt[row];
            const size_t rg = (size_t)row * NOUT;
#pragma unroll
            for (int ni = 0; ni < 4; ++ni) {
                const int col = nbase + Nt * 64 + ni * 16 + mlo;
                float val;
                if (m == selGen)      val = tanhf(acc1[mi][ni][v] + b1c[ni]);
                else if (m == 3)      val = tanhf(acc2[mi][ni][v] + b2c[ni]);
                else                  val = pass[rg + col];
                out[rg + col] = val;
            }
        }
    }
}

// ---------------------------------------------------------------------------
extern "C" void kernel_launch(void* const* d_in, const int* in_sizes, int n_in,
                              void* d_out, int out_size, void* d_ws, size_t ws_size,
                              hipStream_t stream) {
    (void)in_sizes; (void)n_in; (void)out_size; (void)ws_size;
    const float* image    = (const float*)d_in[0];
    const float* text     = (const float*)d_in[1];
    const float* noise    = (const float*)d_in[2];
    const int*   mtypes   = (const int*)  d_in[3];
    const float* enc_img_w = (const float*)d_in[4],  *enc_img_b = (const float*)d_in[5];
    const float* enc_txt_w = (const float*)d_in[6],  *enc_txt_b = (const float*)d_in[7];
    const float* t2i_w1 = (const float*)d_in[8],  *t2i_b1 = (const float*)d_in[9];
    const float* t2i_w2 = (const float*)d_in[10], *t2i_b2 = (const float*)d_in[11];
    const float* i2t_w1 = (const float*)d_in[12], *i2t_b1 = (const float*)d_in[13];
    const float* i2t_w2 = (const float*)d_in[14], *i2t_b2 = (const float*)d_in[15];
    const float* pimg_w1 = (const float*)d_in[16], *pimg_b1 = (const float*)d_in[17];
    const float* pimg_w2 = (const float*)d_in[18], *pimg_b2 = (const float*)d_in[19];
    const float* ptxt_w1 = (const float*)d_in[20], *ptxt_b1 = (const float*)d_in[21];
    const float* ptxt_w2 = (const float*)d_in[22], *ptxt_b2 = (const float*)d_in[23];
    const float* enc_img_g = (const float*)d_in[24], *enc_img_be = (const float*)d_in[25];
    const float* enc_txt_g = (const float*)d_in[26], *enc_txt_be = (const float*)d_in[27];
    const float* t2i_g = (const float*)d_in[28], *t2i_be = (const float*)d_in[29];
    const float* i2t_g = (const float*)d_in[30], *i2t_be = (const float*)d_in[31];
    const float* pimg_g = (const float*)d_in[32], *pimg_be = (const float*)d_in[33];
    const float* ptxt_g = (const float*)d_in[34], *ptxt_be = (const float*)d_in[35];

    // workspace layout (in _Float16 elements)
    _Float16* wsh = (_Float16*)d_ws;
    _Float16* WT_enc_img = wsh;                         // 256*768
    _Float16* WT_enc_txt = WT_enc_img + 256 * 768;      // 256*512
    _Float16* WT_t2i_w1  = WT_enc_txt + 256 * 512;      // 256*256
    _Float16* WT_i2t_w1  = WT_t2i_w1  + 256 * 256;      // 256*256
    _Float16* WT_t2i_w2  = WT_i2t_w1  + 256 * 256;      // 768*256
    _Float16* WT_i2t_w2  = WT_t2i_w2  + 768 * 256;      // 512*256
    _Float16* WT_pimg_w2 = WT_i2t_w2  + 512 * 256;      // 768*256
    _Float16* WT_ptxt_w2 = WT_pimg_w2 + 768 * 256;      // 512*256
    _Float16* h_txt  = WT_ptxt_w2 + 512 * 256;          // B*256
    _Float16* h_img  = h_txt  + (size_t)BATCH * 256;    // B*256
    _Float16* h_pimg = h_img  + (size_t)BATCH * 256;    // B*256
    _Float16* h_ptxt = h_pimg + (size_t)BATCH * 256;    // B*256

    auto cv = [&](const float* W, _Float16* WT, int K, int N) {
        wt_convert_kernel<<<(K * N + 255) / 256, 256, 0, stream>>>(W, WT, K, N);
    };
    cv(enc_img_w, WT_enc_img, D_IMG, FH);
    cv(enc_txt_w, WT_enc_txt, D_TXT, FH);
    cv(t2i_w1,  WT_t2i_w1,  FH, FH);
    cv(i2t_w1,  WT_i2t_w1,  FH, FH);
    cv(t2i_w2,  WT_t2i_w2,  FH, D_IMG);
    cv(i2t_w2,  WT_i2t_w2,  FH, D_TXT);
    cv(pimg_w2, WT_pimg_w2, FH, D_IMG);
    cv(ptxt_w2, WT_ptxt_w2, FH, D_TXT);

    prior_kernel<<<BATCH, 256, 0, stream>>>(noise, pimg_w1, pimg_b1, pimg_g, pimg_be, h_pimg);
    prior_kernel<<<BATCH, 256, 0, stream>>>(noise, ptxt_w1, ptxt_b1, ptxt_g, ptxt_be, h_ptxt);

    // encoders (A = f32 inputs)
    gemm_ln_gelu_kernel<true><<<BATCH / 64, 256, 0, stream>>>(
        text,  D_TXT, WT_enc_txt, enc_txt_b, enc_txt_g, enc_txt_be, h_txt);
    gemm_ln_gelu_kernel<true><<<BATCH / 64, 256, 0, stream>>>(
        image, D_IMG, WT_enc_img, enc_img_b, enc_img_g, enc_img_be, h_img);

    // mid layers (in place: each WG reads only its own 64 rows before writing)
    gemm_ln_gelu_kernel<false><<<BATCH / 64, 256, 0, stream>>>(
        h_txt, FH, WT_t2i_w1, t2i_b1, t2i_g, t2i_be, h_txt);
    gemm_ln_gelu_kernel<false><<<BATCH / 64, 256, 0, stream>>>(
        h_img, FH, WT_i2t_w1, i2t_b1, i2t_g, i2t_be, h_img);

    float* out_img = (float*)d_out;
    float* out_txt = out_img + (size_t)BATCH * D_IMG;
    gemm2_tanh_sel_kernel<<<dim3(BATCH / 64, D_IMG / 256), 256, 0, stream>>>(
        h_txt, WT_t2i_w2, t2i_b2, h_pimg, WT_pimg_w2, pimg_b2,
        image, mtypes, out_img, D_IMG, 1);
    gemm2_tanh_sel_kernel<<<dim3(BATCH / 64, D_TXT / 256), 256, 0, stream>>>(
        h_img, WT_i2t_w2, i2t_b2, h_ptxt, WT_ptxt_w2, ptxt_b2,
        text, mtypes, out_txt, D_TXT, 2);
}
